// MultiHeadAttentionBlock_20401094656723
// MI455X (gfx1250) — matrix operand order repaired and checked by reading, not compiled, not run on hardware
//
#include <hip/hip_runtime.h>

// ---------------------------------------------------------------------------
// MI455X (gfx1250) multi-head attention:
//   - all matmuls on v_wmma_f32_16x16x32_bf16 matrix cores (fp32 accumulate)
//   - flash-attention K/V tiles staged via the Tensor Data Mover
//     (tensor_load_to_lds + s_wait_tensorcnt), with D# LDS padding giving a
//     bank-conflict-free 72-element LDS row stride
// ---------------------------------------------------------------------------

typedef __attribute__((ext_vector_type(16))) __bf16        v16bf;
typedef __attribute__((ext_vector_type(8)))  float          v8f;
typedef __attribute__((ext_vector_type(8)))  unsigned short v8us;
typedef __attribute__((ext_vector_type(4)))  unsigned int   v4u;
typedef __attribute__((ext_vector_type(8)))  int            v8i;
typedef __attribute__((ext_vector_type(4)))  int            v4i;
typedef unsigned short bf16_t;

#define EMB 1024
#define SEQ 2048
#define NB  4
#define NH  16
#define DH  64

union Frag { v16bf v; v8us h[2]; };

// fp32 -> bf16 round-to-nearest-even
__device__ __forceinline__ bf16_t f2bf(float f) {
  unsigned int u = __float_as_uint(f);
  u += 0x7FFFu + ((u >> 16) & 1u);
  return (bf16_t)(u >> 16);
}

__device__ __forceinline__ v8f wmma_bf16(const Frag& a, const Frag& b, v8f c) {
  return __builtin_amdgcn_wmma_f32_16x16x32_bf16(false, a.v, false, b.v,
                                                 (short)0, c, false, false);
}

// ---------------------------------------------------------------------------
// Tensor Data Mover: 2-D tile load (bf16 elements) global -> LDS.
// LDS padding: pad_interval=4 (every 32 DWORDs = 64 elems) + pad_amount=3
// (4 DWORDs = 8 elems) => effective LDS row stride 72 elements.
// D# packing per CDNA5 ISA sec 8.3/8.4.  Issued once per wave; EXEC ignored.
// This toolchain exposes the 6-arg builtin:
//   (uint32x4 g0, int32x8 g1, int32x4 g2, int32x4 g3, int32x8, i32 cpol)
// ---------------------------------------------------------------------------
__device__ __forceinline__ void tdm_load_tile_2d(unsigned lds_addr,
                                                 const bf16_t* gaddr,
                                                 unsigned tile_w,   // elems
                                                 unsigned tile_h,   // rows
                                                 unsigned stride) { // elems
  unsigned long long ga = (unsigned long long)(uintptr_t)gaddr;
  v4u g0 = { 1u,                                  // count=1, user mode
             lds_addr,                            // LDS byte address
             (unsigned)(ga & 0xFFFFFFFFu),        // global_addr[31:0]
             (unsigned)((ga >> 32) & 0x1FFFFFFu)  // global_addr[56:32]
               | (2u << 30) };                    // type=2 ("image")
  v8i g1;
  g1[0] = (int)((1u << 16)      // data_size = 1 -> 2-byte elements
              | (1u << 20)      // pad_enable
              | (4u << 22)      // pad_interval: 32 DWORDs
              | (3u << 25));    // pad_amount: 4 DWORDs
  g1[1] = (int)(tile_w << 16);                        // tensor_dim0[15:0]
  g1[2] = (int)((tile_w >> 16) | (tile_h << 16));     // dim0 hi | dim1 lo
  g1[3] = (int)((tile_h >> 16) | (tile_w << 16));     // dim1 hi | tile_dim0
  g1[4] = (int)tile_h;                                // tile_dim1 (tile_dim2=0)
  g1[5] = (int)stride;                                // dim0_stride[31:0]
  g1[6] = 0;                                          // dim0_stride hi | dim1_stride lo
  g1[7] = 0;
  v4i gz4 = {0, 0, 0, 0};
  v8i gz8 = {0, 0, 0, 0, 0, 0, 0, 0};
  __builtin_amdgcn_tensor_load_to_lds(g0, g1, gz4, gz4, gz8, 0);
}

// low 32 bits of a generic LDS address = workgroup-relative LDS byte offset
__device__ __forceinline__ unsigned lds_off(const void* p) {
  return (unsigned)(uintptr_t)p;
}

// ---------------------------------------------------------------------------
// Kernel 1: fused Q/K/V projection.  Y = X @ W^T   (M=8192, N=1024, K=1024)
// blockDim=256 (8 waves), tile 128x128, BK=32.  grid.z: 0=Q, 1=K, 2=V.
// Q/K stored [B,H,S,d] bf16 (Q pre-scaled by 1/sqrt(d)); V stored [B,H,d,S].
// ---------------------------------------------------------------------------
__global__ __launch_bounds__(256) void qkv_proj_kernel(
    const float* __restrict__ Xq, const float* __restrict__ Xk,
    const float* __restrict__ Xv,
    const float* __restrict__ Wq, const float* __restrict__ Wk,
    const float* __restrict__ Wv,
    bf16_t* __restrict__ qout, bf16_t* __restrict__ kout,
    bf16_t* __restrict__ vtout)
{
  __shared__ bf16_t As[128][40];   // 40-stride pad: 80B rows, 16B aligned
  __shared__ bf16_t Bs[128][40];

  const int mode = blockIdx.z;
  const float* X = (mode == 0) ? Xq : (mode == 1) ? Xk : Xv;
  const float* W = (mode == 0) ? Wq : (mode == 1) ? Wk : Wv;

  const int m0 = blockIdx.y * 128;
  const int n0 = blockIdx.x * 128;
  const int tid  = threadIdx.x;
  const int wid  = tid >> 5, lane = tid & 31;
  const int l16  = lane & 15, hi = lane >> 4;
  const int wm   = wid & 3,  wn = wid >> 2;      // waves: 4 (M) x 2 (N)
  const int crow = tid >> 1, chalf = tid & 1;    // cooperative staging

  v8f acc[2][4];
  #pragma unroll
  for (int i = 0; i < 2; ++i)
    #pragma unroll
    for (int j = 0; j < 4; ++j) acc[i][j] = 0.0f;

  const float* sa_base = X + (size_t)(m0 + crow) * EMB + chalf * 16;
  const float* sb_base = W + (size_t)(n0 + crow) * EMB + chalf * 16;

  for (int k0 = 0; k0 < EMB; k0 += 32) {
    // stage A (X rows) and B (W rows == columns of B=K x N) as bf16
    {
      const float4* sa = (const float4*)(sa_base + k0);
      const float4* sb = (const float4*)(sb_base + k0);
      if (k0 + 32 < EMB) {                 // prefetch next K chunk
        __builtin_prefetch(sa_base + k0 + 32);
        __builtin_prefetch(sb_base + k0 + 32);
      }
      #pragma unroll
      for (int q = 0; q < 4; ++q) {
        float4 a4 = sa[q], b4 = sb[q];
        int c = chalf * 16 + q * 4;
        As[crow][c+0]=f2bf(a4.x); As[crow][c+1]=f2bf(a4.y);
        As[crow][c+2]=f2bf(a4.z); As[crow][c+3]=f2bf(a4.w);
        Bs[crow][c+0]=f2bf(b4.x); Bs[crow][c+1]=f2bf(b4.y);
        Bs[crow][c+2]=f2bf(b4.z); Bs[crow][c+3]=f2bf(b4.w);
      }
    }
    __syncthreads();

    Frag af[2], bf[4];
    #pragma unroll
    for (int tm = 0; tm < 2; ++tm) {            // A: lane = row, K chunks per ISA
      const bf16_t* p = &As[wm*32 + tm*16 + l16][0];
      const int c0 = hi ? 8 : 0;
      af[tm].h[0] = *(const v8us*)(p + c0);
      af[tm].h[1] = *(const v8us*)(p + c0 + 16);
    }
    #pragma unroll
    for (int tn = 0; tn < 4; ++tn) {            // B: lane = col, K chunks per ISA
      const bf16_t* p = &Bs[wn*64 + tn*16 + l16][0];
      bf[tn].h[0] = *(const v8us*)(p + hi*8);
      bf[tn].h[1] = *(const v8us*)(p + hi*8 + 16);
    }
    #pragma unroll
    for (int tm = 0; tm < 2; ++tm)
      #pragma unroll
      for (int tn = 0; tn < 4; ++tn)
        acc[tm][tn] = wmma_bf16(af[tm], bf[tn], acc[tm][tn]);
    __syncthreads();
  }

  // epilogue: mode branch hoisted out of the store loops
  if (mode == 2) {
    #pragma unroll
    for (int tm = 0; tm < 2; ++tm)
      #pragma unroll
      for (int tn = 0; tn < 4; ++tn)
        #pragma unroll
        for (int j = 0; j < 8; ++j) {
          const int gm = m0 + wm*32 + tm*16 + j + 8*hi;
          const int gn = n0 + wn*64 + tn*16 + l16;
          const int b = gm >> 11, s = gm & (SEQ - 1);
          const int h = gn >> 6,  dh = gn & (DH - 1);
          vtout[(((size_t)b*NH + h)*DH + dh)*SEQ + s] = f2bf(acc[tm][tn][j]);
        }
  } else {
    bf16_t* outp = (mode == 0) ? qout : kout;
    const float sc = (mode == 0) ? 0.125f : 1.0f;   // fold 1/sqrt(DH) into Q
    #pragma unroll
    for (int tm = 0; tm < 2; ++tm)
      #pragma unroll
      for (int tn = 0; tn < 4; ++tn)
        #pragma unroll
        for (int j = 0; j < 8; ++j) {
          const int gm = m0 + wm*32 + tm*16 + j + 8*hi;
          const int gn = n0 + wn*64 + tn*16 + l16;
          const int b = gm >> 11, s = gm & (SEQ - 1);
          const int h = gn >> 6,  dh = gn & (DH - 1);
          outp[(((size_t)b*NH + h)*SEQ + s)*DH + dh] = f2bf(acc[tm][tn][j] * sc);
        }
  }
}

// ---------------------------------------------------------------------------
// Kernel 2: causal flash attention.  blockDim=128 (4 waves), 64 query rows
// per block (16 per wave).  K/V 64x64 bf16 tiles staged by the TDM with LDS
// padding (stride 72).  grid = (SEQ/64, B*H).
// ---------------------------------------------------------------------------
__global__ __launch_bounds__(128) void flash_kernel(
    const bf16_t* __restrict__ qb, const bf16_t* __restrict__ kb,
    const bf16_t* __restrict__ vtb, bf16_t* __restrict__ attnb)
{
  __shared__ bf16_t Kt[64][72];      // [key][d]   (TDM pad -> stride 72)
  __shared__ bf16_t Vt[64][72];      // [d][key]   (from V^T)
  __shared__ bf16_t Pt[4][16][72];   // per-wave P transpose bounce

  const int qblk = blockIdx.x;
  const int bh   = blockIdx.y;
  const int tid  = threadIdx.x;
  const int w    = tid >> 5, lane = tid & 31;
  const int l16  = lane & 15, hi = lane >> 4;

  const bf16_t* qh = qb  + (size_t)bh * SEQ * DH;
  const bf16_t* kh = kb  + (size_t)bh * SEQ * DH;
  const bf16_t* vh = vtb + (size_t)bh * DH * SEQ;

  const unsigned kt_lds = lds_off(&Kt[0][0]);
  const unsigned vt_lds = lds_off(&Vt[0][0]);

  // Q fragments for this wave's 16 rows (d = 2 chunks of 32)
  Frag qf[2];
  {
    const bf16_t* qp = qh + (size_t)(qblk*64 + w*16 + l16) * DH;
    #pragma unroll
    for (int kc = 0; kc < 2; ++kc) {
      const int c0 = kc*32 + (hi ? 8 : 0);
      qf[kc].h[0] = *(const v8us*)(qp + c0);
      qf[kc].h[1] = *(const v8us*)(qp + c0 + 16);
    }
  }

  v8f O[4];
  #pragma unroll
  for (int t = 0; t < 4; ++t) O[t] = 0.0f;
  float mrow[8], lrow[8];
  #pragma unroll
  for (int j = 0; j < 8; ++j) { mrow[j] = -1e30f; lrow[j] = 0.0f; }

  for (int kv = 0; kv <= qblk; ++kv) {
    __syncthreads();   // previous iteration's LDS readers are done
    if (w == 0) {      // wave 0 drives the Tensor Data Mover
      tdm_load_tile_2d(kt_lds, kh + (size_t)(kv*64) * DH, DH, 64, DH);
      tdm_load_tile_2d(vt_lds, vh + kv*64,                64, DH, SEQ);
      __builtin_amdgcn_s_wait_tensorcnt(0);
    }
    __syncthreads();   // tiles visible to all waves

    // S = Q K^T  (16 queries x 64 keys), fp32 accum
    v8f st[4];
    #pragma unroll
    for (int tn = 0; tn < 4; ++tn) {
      v8f s = 0.0f;
      #pragma unroll
      for (int kc = 0; kc < 2; ++kc) {
        Frag kf;
        const bf16_t* p = &Kt[tn*16 + l16][kc*32 + hi*8];
        kf.h[0] = *(const v8us*)(p);
        kf.h[1] = *(const v8us*)(p + 16);
        s = wmma_bf16(qf[kc], kf, s);
      }
      st[tn] = s;
    }

    // causal mask only on the diagonal KV block
    if (kv == qblk) {
      #pragma unroll
      for (int tn = 0; tn < 4; ++tn)
        #pragma unroll
        for (int j = 0; j < 8; ++j)
          if (tn*16 + l16 > w*16 + j + 8*hi) st[tn][j] = -1e30f;
    }

    // online softmax: row reductions across the 16-lane half-groups
    float alpha[8];
    #pragma unroll
    for (int j = 0; j < 8; ++j) {
      float mx = fmaxf(fmaxf(st[0][j], st[1][j]), fmaxf(st[2][j], st[3][j]));
      #pragma unroll
      for (int d = 1; d < 16; d <<= 1) mx = fmaxf(mx, __shfl_xor(mx, d, 32));
      const float mnew = fmaxf(mrow[j], mx);
      alpha[j] = __expf(mrow[j] - mnew);
      float r = 0.0f;
      #pragma unroll
      for (int tn = 0; tn < 4; ++tn) {
        const float p = __expf(st[tn][j] - mnew);
        st[tn][j] = p;
        r += p;
      }
      #pragma unroll
      for (int d = 1; d < 16; d <<= 1) r += __shfl_xor(r, d, 32);
      lrow[j] = lrow[j] * alpha[j] + r;
      mrow[j] = mnew;
    }
    #pragma unroll
    for (int t = 0; t < 4; ++t)
      #pragma unroll
      for (int j = 0; j < 8; ++j) O[t][j] *= alpha[j];

    // P: C-layout -> row-major LDS -> A-fragment layout
    #pragma unroll
    for (int t = 0; t < 4; ++t)
      #pragma unroll
      for (int j = 0; j < 8; ++j)
        Pt[w][j + 8*hi][t*16 + l16] = f2bf(st[t][j]);
    __syncthreads();

    Frag pf[2];
    #pragma unroll
    for (int kc = 0; kc < 2; ++kc) {
      const bf16_t* p = &Pt[w][l16][0];
      const int c0 = kc*32 + (hi ? 8 : 0);
      pf[kc].h[0] = *(const v8us*)(p + c0);
      pf[kc].h[1] = *(const v8us*)(p + c0 + 16);
    }
    // O += P V  (keys = 2 chunks of 32, dh = 4 groups of 16)
    #pragma unroll
    for (int t = 0; t < 4; ++t) {
      #pragma unroll
      for (int kc = 0; kc < 2; ++kc) {
        Frag vf;
        const bf16_t* p = &Vt[t*16 + l16][kc*32 + hi*8];
        vf.h[0] = *(const v8us*)(p);
        vf.h[1] = *(const v8us*)(p + 16);
        O[t] = wmma_bf16(pf[kc], vf, O[t]);
      }
    }
  }

  // normalize + store merged-head bf16 [B,S,E]
  const int b = bh >> 4, h = bh & 15;
  #pragma unroll
  for (int j = 0; j < 8; ++j) {
    const float inv = 1.0f / lrow[j];
    const int sgl = qblk*64 + w*16 + j + 8*hi;
    #pragma unroll
    for (int t = 0; t < 4; ++t) {
      const int col = h*64 + t*16 + l16;
      attnb[((size_t)b*SEQ + sgl)*EMB + col] = f2bf(O[t][j] * inv);
    }
  }
}

// ---------------------------------------------------------------------------
// Kernel 3: output projection.  out = attn @ Wo^T   (fp32 store)
// ---------------------------------------------------------------------------
__global__ __launch_bounds__(256) void out_proj_kernel(
    const bf16_t* __restrict__ A, const float* __restrict__ Wo,
    float* __restrict__ out)
{
  __shared__ bf16_t As[128][40];
  __shared__ bf16_t Bs[128][40];

  const int m0 = blockIdx.y * 128;
  const int n0 = blockIdx.x * 128;
  const int tid  = threadIdx.x;
  const int wid  = tid >> 5, lane = tid & 31;
  const int l16  = lane & 15, hi = lane >> 4;
  const int wm   = wid & 3,  wn = wid >> 2;
  const int crow = tid >> 1, chalf = tid & 1;

  v8f acc[2][4];
  #pragma unroll
  for (int i = 0; i < 2; ++i)
    #pragma unroll
    for (int j = 0; j < 4; ++j) acc[i][j] = 0.0f;

  const bf16_t* sa_base = A + (size_t)(m0 + crow) * EMB + chalf * 16;
  const float*  sb_base = Wo + (size_t)(n0 + crow) * EMB + chalf * 16;

  for (int k0 = 0; k0 < EMB; k0 += 32) {
    {
      const v8us*   sa = (const v8us*)(sa_base + k0);
      const float4* sb = (const float4*)(sb_base + k0);
      if (k0 + 32 < EMB) {
        __builtin_prefetch(sa_base + k0 + 32);
        __builtin_prefetch(sb_base + k0 + 32);
      }
      *(v8us*)&As[crow][chalf*16]     = sa[0];
      *(v8us*)&As[crow][chalf*16 + 8] = sa[1];
      #pragma unroll
      for (int q = 0; q < 4; ++q) {
        float4 b4 = sb[q];
        int c = chalf * 16 + q * 4;
        Bs[crow][c+0]=f2bf(b4.x); Bs[crow][c+1]=f2bf(b4.y);
        Bs[crow][c+2]=f2bf(b4.z); Bs[crow][c+3]=f2bf(b4.w);
      }
    }
    __syncthreads();

    Frag af[2], bf[4];
    #pragma unroll
    for (int tm = 0; tm < 2; ++tm) {
      const bf16_t* p = &As[wm*32 + tm*16 + l16][0];
      const int c0 = hi ? 8 : 0;
      af[tm].h[0] = *(const v8us*)(p + c0);
      af[tm].h[1] = *(const v8us*)(p + c0 + 16);
    }
    #pragma unroll
    for (int tn = 0; tn < 4; ++tn) {
      const bf16_t* p = &Bs[wn*64 + tn*16 + l16][0];
      bf[tn].h[0] = *(const v8us*)(p + hi*8);
      bf[tn].h[1] = *(const v8us*)(p + hi*8 + 16);
    }
    #pragma unroll
    for (int tm = 0; tm < 2; ++tm)
      #pragma unroll
      for (int tn = 0; tn < 4; ++tn)
        acc[tm][tn] = wmma_bf16(af[tm], bf[tn], acc[tm][tn]);
    __syncthreads();
  }

  #pragma unroll
  for (int tm = 0; tm < 2; ++tm)
    #pragma unroll
    for (int tn = 0; tn < 4; ++tn)
      #pragma unroll
      for (int j = 0; j < 8; ++j) {
        const int gm = m0 + wm*32 + tm*16 + j + 8*hi;
        const int gn = n0 + wn*64 + tn*16 + l16;
        out[(size_t)gm * EMB + gn] = acc[tm][tn][j];
      }
}

// ---------------------------------------------------------------------------
extern "C" void kernel_launch(void* const* d_in, const int* in_sizes, int n_in,
                              void* d_out, int out_size, void* d_ws, size_t ws_size,
                              hipStream_t stream) {
  (void)in_sizes; (void)n_in; (void)out_size; (void)ws_size;
  const float* key   = (const float*)d_in[0];
  const float* query = (const float*)d_in[1];
  const float* value = (const float*)d_in[2];
  // d_in[3] = mask: causal tril by construction; implemented directly.
  const float* Wq = (const float*)d_in[4];
  const float* Wk = (const float*)d_in[5];
  const float* Wv = (const float*)d_in[6];
  const float* Wo = (const float*)d_in[7];
  float* out = (float*)d_out;

  const size_t NE = (size_t)NB * NH * SEQ * DH;   // 8.4M elements per buffer
  bf16_t* qbuf  = (bf16_t*)d_ws;
  bf16_t* kbuf  = qbuf + NE;
  bf16_t* vtbuf = kbuf + NE;
  bf16_t* attnb = vtbuf + NE;                     // total ~67 MB workspace

  qkv_proj_kernel<<<dim3(EMB/128, (NB*SEQ)/128, 3), 256, 0, stream>>>(
      query, key, value, Wq, Wk, Wv, qbuf, kbuf, vtbuf);
  flash_kernel<<<dim3(SEQ/64, NB*NH), 128, 0, stream>>>(qbuf, kbuf, vtbuf, attnb);
  out_proj_kernel<<<dim3(EMB/128, (NB*SEQ)/128), 256, 0, stream>>>(attnb, Wo, out);
}
